// GNNModel_62569083568519
// MI455X (gfx1250) — matrix operand order repaired and checked by reading, not compiled, not run on hardware
//
#include <hip/hip_runtime.h>
#include <hip/hip_bf16.h>
#include <math.h>

typedef __attribute__((ext_vector_type(16))) _Float16 v16h;
typedef __attribute__((ext_vector_type(8)))  _Float16 v8h;
typedef __attribute__((ext_vector_type(8)))  float    v8f;

#define NN 100000
#define NE 3200000
#define IN_DIM 256
#define HID 64
#define HID2 32

// ---------------------------------------------------------------------------
// Degree / normalization
// ---------------------------------------------------------------------------
__global__ void init_deg(float* __restrict__ deg) {
    int i = blockIdx.x * blockDim.x + threadIdx.x;
    if (i < NN) deg[i] = 1.0f;   // self loop
}

__global__ void accum_deg(const int* __restrict__ dst, float* __restrict__ deg) {
    int e = blockIdx.x * blockDim.x + threadIdx.x;
    if (e < NE) atomicAdd(&deg[dst[e]], 1.0f);
}

__global__ void deg_to_dinv(float* __restrict__ deg) {
    int i = blockIdx.x * blockDim.x + threadIdx.x;
    if (i < NN) {
        float d = deg[i];
        deg[i] = (d > 0.0f) ? rsqrtf(d) : 0.0f;
    }
}

// ---------------------------------------------------------------------------
// WMMA GEMM: C[M,N] = A[M,K] * W[K,N], fp32 in/out, f16 operands, f32 acc.
// W is staged once per block into LDS, f16, column-major with +8 half pad
// (conflict-free ds_load_b128 fragments). One wave computes a 16-row x N
// strip so A (the big operand) is streamed from HBM exactly once.
// M % 16 == 0, K % 32 == 0, N % 16 == 0.
// ---------------------------------------------------------------------------
template<int K, int N>
__global__ void gemm_wmma(const float* __restrict__ A, const float* __restrict__ W,
                          float* __restrict__ C, int mtiles) {
    constexpr int KP = K + 8;                 // padded LDS row pitch (halves)
    __shared__ _Float16 wlds[N * KP];

    // ---- cooperative stage: wlds[col*KP + k] = f16(W[k*N + col])
    for (int idx = threadIdx.x; idx < K * N; idx += blockDim.x) {
        int k = idx / N, col = idx - k * N;   // coalesced global read
        wlds[col * KP + k] = (_Float16)W[idx];
    }
    __syncthreads();

    const int wave = threadIdx.x >> 5;
    const int lane = threadIdx.x & 31;
    const int mt   = blockIdx.x * (blockDim.x >> 5) + wave;
    if (mt >= mtiles) return;                 // wave-uniform: EXEC all-ones inside

    const int grp = lane >> 4;                // 0: lanes 0-15, 1: lanes 16-31
    const int l16 = lane & 15;
    constexpr int NT = N / 16;

    v8f zero = {};
    v8f acc[NT];
#pragma unroll
    for (int nt = 0; nt < NT; ++nt) acc[nt] = zero;

    const float* arow = A + (size_t)(mt * 16 + l16) * K;   // this lane's A row

    for (int k0 = 0; k0 < K; k0 += 32) {
        // --- A fragment (16x32 f16): lane(row=l16) holds K = k0+grp*8 .. +7
        //     in elems 0..7 and K = k0+16+grp*8 .. +7 in elems 8..15
        const float4* ap = (const float4*)(arow + k0 + grp * 8);
        float4 f0 = ap[0], f1 = ap[1];        // K offsets +0..7
        float4 f2 = ap[4], f3 = ap[5];        // K offsets +16..23
        v16h a;
        a[0]  = (_Float16)f0.x; a[1]  = (_Float16)f0.y;
        a[2]  = (_Float16)f0.z; a[3]  = (_Float16)f0.w;
        a[4]  = (_Float16)f1.x; a[5]  = (_Float16)f1.y;
        a[6]  = (_Float16)f1.z; a[7]  = (_Float16)f1.w;
        a[8]  = (_Float16)f2.x; a[9]  = (_Float16)f2.y;
        a[10] = (_Float16)f2.z; a[11] = (_Float16)f2.w;
        a[12] = (_Float16)f3.x; a[13] = (_Float16)f3.y;
        a[14] = (_Float16)f3.z; a[15] = (_Float16)f3.w;

        const int kbase = k0 + grp * 8;       // B: lanes 16-31 hold K=+8..15,+24..31
#pragma unroll
        for (int nt = 0; nt < NT; ++nt) {
            const int col = nt * 16 + l16;
            const _Float16* bp = &wlds[col * KP + kbase];  // 16B-aligned
            v8h blo = *(const v8h*)bp;
            v8h bhi = *(const v8h*)(bp + 16);
            v16h b = __builtin_shufflevector(blo, bhi,
                0, 1, 2, 3, 4, 5, 6, 7, 8, 9, 10, 11, 12, 13, 14, 15);
            acc[nt] = __builtin_amdgcn_wmma_f32_16x16x32_f16(
                false, a, false, b, (short)0, acc[nt], false, false);
        }
    }
    // --- store D: VGPR v -> row grp*8+v, col = nt*16+l16
#pragma unroll
    for (int nt = 0; nt < NT; ++nt) {
        const int col = nt * 16 + l16;
#pragma unroll
        for (int v = 0; v < 8; ++v) {
            const int r = mt * 16 + grp * 8 + v;
            C[(size_t)r * N + col] = acc[nt][v];
        }
    }
}

// ---------------------------------------------------------------------------
// Propagation: p[i,:] = dinv[i]^2 * h[i,:]  (self loop), then per-edge
// p[dst,:] += dinv[src]*dinv[dst] * h[src,:] via L2-resident f32 atomics.
// ---------------------------------------------------------------------------
template<int F>
__global__ void self_loop_init(const float* __restrict__ h,
                               const float* __restrict__ dinv,
                               float* __restrict__ p) {
    int t = blockIdx.x * blockDim.x + threadIdx.x;       // over NN*F/4
    if (t >= NN * (F / 4)) return;
    int n = t / (F / 4);
    float w = dinv[n]; w *= w;
    float4 hv = ((const float4*)h)[t];
    float4 o = { hv.x * w, hv.y * w, hv.z * w, hv.w * w };
    ((float4*)p)[t] = o;
}

template<int F>
__global__ void scatter_edges(const int* __restrict__ src, const int* __restrict__ dst,
                              const float* __restrict__ h, const float* __restrict__ dinv,
                              float* __restrict__ p) {
    constexpr int C4 = F / 4;
    long long t = (long long)blockIdx.x * blockDim.x + threadIdx.x;  // NE*C4 threads
    if (t >= (long long)NE * C4) return;
    int e = (int)(t / C4);
    int j = (int)(t % C4);
    int s = src[e], d = dst[e];
    float norm = dinv[s] * dinv[d];
    float4 hv = ((const float4*)(h + (size_t)s * F))[j];
    float* o = p + (size_t)d * F + j * 4;
    atomicAdd(o + 0, hv.x * norm);
    atomicAdd(o + 1, hv.y * norm);
    atomicAdd(o + 2, hv.z * norm);
    atomicAdd(o + 3, hv.w * norm);
}

template<int F>
__global__ void bias_relu(float* __restrict__ p, const float* __restrict__ b) {
    int t = blockIdx.x * blockDim.x + threadIdx.x;       // over NN*F
    if (t >= NN * F) return;
    float v = p[t] + b[t & (F - 1)];
    p[t] = v > 0.0f ? v : 0.0f;
}

// ---------------------------------------------------------------------------
// Final head: out[n] = sigmoid(dot(a2[n, 0:32], fc_w) + fc_b); one wave/node
// ---------------------------------------------------------------------------
__global__ void final_mlp(const float* __restrict__ a2, const float* __restrict__ fcw,
                          const float* __restrict__ fcb, float* __restrict__ out) {
    int wave = threadIdx.x >> 5;
    int lane = threadIdx.x & 31;
    int n = blockIdx.x * (blockDim.x >> 5) + wave;
    if (n >= NN) return;
    float v = a2[(size_t)n * HID2 + lane] * fcw[lane];
#pragma unroll
    for (int off = 16; off > 0; off >>= 1)
        v += __shfl_xor(v, off, 32);
    if (lane == 0) out[n] = 1.0f / (1.0f + expf(-(v + fcb[0])));
}

// ---------------------------------------------------------------------------
extern "C" void kernel_launch(void* const* d_in, const int* in_sizes, int n_in,
                              void* d_out, int out_size, void* d_ws, size_t ws_size,
                              hipStream_t stream) {
    const float* x   = (const float*)d_in[0];
    const int*   edg = (const int*)d_in[1];      // [2, NE] row-major
    const int*   src = edg;
    const int*   dst = edg + NE;
    const float* W1  = (const float*)d_in[2];
    const float* b1  = (const float*)d_in[3];
    const float* W2  = (const float*)d_in[4];
    const float* b2  = (const float*)d_in[5];
    const float* fcw = (const float*)d_in[6];
    const float* fcb = (const float*)d_in[7];
    float* out = (float*)d_out;

    float* ws   = (float*)d_ws;
    float* dinv = ws;                               // NN
    float* h1   = dinv + NN;                        // NN*HID
    float* p1   = h1 + (size_t)NN * HID;            // NN*HID
    float* h2   = p1 + (size_t)NN * HID;            // NN*HID2
    float* p2   = h2 + (size_t)NN * HID2;           // NN*HID2

    const int B = 256;
    const int mtiles = NN / 16;                     // 6250 exactly

    // normalization coefficients
    init_deg   <<<(NN + B - 1) / B, B, 0, stream>>>(dinv);
    accum_deg  <<<(NE + B - 1) / B, B, 0, stream>>>(dst, dinv);
    deg_to_dinv<<<(NN + B - 1) / B, B, 0, stream>>>(dinv);

    // layer 1: h1 = x @ W1 ; p1 = Ahat h1 ; a1 = relu(p1 + b1)
    gemm_wmma<IN_DIM, HID><<<(mtiles + 7) / 8, B, 0, stream>>>(x, W1, h1, mtiles);
    self_loop_init<HID><<<(NN * (HID / 4) + B - 1) / B, B, 0, stream>>>(h1, dinv, p1);
    scatter_edges<HID><<<(int)(((long long)NE * (HID / 4) + B - 1) / B), B, 0, stream>>>(
        src, dst, h1, dinv, p1);
    bias_relu<HID><<<(NN * HID + B - 1) / B, B, 0, stream>>>(p1, b1);

    // layer 2: h2 = a1 @ W2 ; p2 = Ahat h2 ; a2 = relu(p2 + b2)
    gemm_wmma<HID, HID2><<<(mtiles + 7) / 8, B, 0, stream>>>(p1, W2, h2, mtiles);
    self_loop_init<HID2><<<(NN * (HID2 / 4) + B - 1) / B, B, 0, stream>>>(h2, dinv, p2);
    scatter_edges<HID2><<<(int)(((long long)NE * (HID2 / 4) + B - 1) / B), B, 0, stream>>>(
        src, dst, h2, dinv, p2);
    bias_relu<HID2><<<(NN * HID2 + B - 1) / B, B, 0, stream>>>(p2, b2);

    // head
    final_mlp<<<(NN + 7) / 8, B, 0, stream>>>(p2, fcw, fcb, out);
}
